// Net_74397423501440
// MI455X (gfx1250) — matrix operand order repaired and checked by reading, not compiled, not run on hardware
//
#include <hip/hip_runtime.h>

#define NNODES 50000
#define NEDGES 1600000
#define FIN    128
#define EMB    48
#define HID    48
#define FOUT   64
#define KORD   7

typedef __attribute__((ext_vector_type(16))) __bf16 v16bf;
typedef __attribute__((ext_vector_type(8)))  float  v8f;

__device__ __forceinline__ __bf16 f2bf(float f) { return (__bf16)f; }

// ---- WMMA operand loaders (layouts per CDNA5 ISA 16-bit A 16x32 / B 32x16 tables) ----

// Full 32-deep A step: lane<16 -> M=lane, K pairs {2v,2v+1} (v<4) / {2v+8,2v+9} (v>=4),
// +8 K-offset for lanes 16-31.
template <class FA>
__device__ __forceinline__ v16bf load_a_full(FA fa, int l15, int half, int kb) {
  v16bf av;
#pragma unroll
  for (int v = 0; v < 8; ++v) {
    int kp = kb + (v < 4 ? 2 * v : 2 * v + 8) + half * 8;
    av[2 * v]     = f2bf(fa(l15, kp));
    av[2 * v + 1] = f2bf(fa(l15, kp + 1));
  }
  return av;
}

// 16-deep tail (K in [kb, kb+16)): exactly the v<4 pairs for BOTH halves -> branch-free.
template <class FA>
__device__ __forceinline__ v16bf load_a_tail16(FA fa, int l15, int half, int kb) {
  v16bf av = {};
#pragma unroll
  for (int v = 0; v < 4; ++v) {
    int kp = kb + 2 * v + half * 8;
    av[2 * v]     = f2bf(fa(l15, kp));
    av[2 * v + 1] = f2bf(fa(l15, kp + 1));
  }
  return av;
}

// B 32x16: lane&15 = N, K order matches A (j<8 -> K=j, j>=8 -> K=j+8, +8 for lanes 16-31);
// fb(k,n) reads a TRANSPOSED (n-major, K-padded) weight array so the two 8-K runs
// per lane are contiguous.
template <class FB>
__device__ __forceinline__ v16bf load_b(FB fb, int l15, int half, int kb) {
  v16bf bv;
  int kof = kb + half * 8;
#pragma unroll
  for (int j = 0; j < 16; ++j) bv[j] = f2bf(fb(kof + (j < 8 ? j : j + 8), l15));
  return bv;
}

// One 16x16 f32 tile, KDIM compile-time; tail handled branch-free.
template <int KDIM, class FA, class FB>
__device__ __forceinline__ v8f wmma_tile(FA fa, FB fb, v8f c, int lane) {
  const int half = lane >> 4;
  const int l15  = lane & 15;
#pragma unroll
  for (int kb = 0; kb < KDIM; kb += 32) {
    v16bf av = (kb + 32 <= KDIM) ? load_a_full(fa, l15, half, kb)
                                 : load_a_tail16(fa, l15, half, kb);
    v16bf bv = load_b(fb, l15, half, kb);
    c = __builtin_amdgcn_wmma_f32_16x16x32_bf16(false, av, false, bv, (short)0, c,
                                                false, false);
  }
  return c;
}

// ---------------- edge-side kernels ----------------

__global__ void k_deg(const int* __restrict__ row, const float* __restrict__ w,
                      float* __restrict__ deg) {
  int e = blockIdx.x * blockDim.x + threadIdx.x;
  if (e < NEDGES) atomicAdd(&deg[row[e]], w[e]);
}

__global__ void k_dinv(const float* __restrict__ deg, float* __restrict__ dinv) {
  int i = blockIdx.x * blockDim.x + threadIdx.x;
  if (i < NNODES) {
    float d = deg[i];
    dinv[i] = d > 0.0f ? rsqrtf(d) : 0.0f;
  }
}

__global__ void k_norm(const int* __restrict__ row, const int* __restrict__ col,
                       const float* __restrict__ w, const float* __restrict__ dinv,
                       float* __restrict__ norm) {
  int e = blockIdx.x * blockDim.x + threadIdx.x;
  if (e < NEDGES) norm[e] = -dinv[row[e]] * w[e] * dinv[col[e]];
}

// S[col[e], :] += norm[e] * T[row[e], :], 12 threads/edge (float4 each)
__global__ void k_prop(const int* __restrict__ row, const int* __restrict__ col,
                       const float* __restrict__ norm, const float* __restrict__ T,
                       float* __restrict__ S) {
  int tid = blockIdx.x * blockDim.x + threadIdx.x;
  if (tid >= NEDGES * 12) return;
  int e = tid / 12;
  int c4 = tid - e * 12;
  int r = row[e], cl = col[e];
  float nv = norm[e];
  const float4 v = *reinterpret_cast<const float4*>(T + (size_t)r * EMB + c4 * 4);
  float* dst = S + (size_t)cl * EMB + c4 * 4;
  atomicAdd(dst + 0, nv * v.x);
  atomicAdd(dst + 1, nv * v.y);
  atomicAdd(dst + 2, nv * v.z);
  atomicAdd(dst + 3, nv * v.w);
}

// Tcur = alpha*S + beta*P  (float4 vectorized; N*EMB % 4 == 0)
__global__ void k_combine(const float4* __restrict__ S, const float4* __restrict__ P,
                          float4* __restrict__ T, float alpha, float beta) {
  int i = blockIdx.x * blockDim.x + threadIdx.x;
  if (i < NNODES * EMB / 4) {
    float4 s = S[i], p = P[i];
    T[i] = make_float4(alpha * s.x + beta * p.x, alpha * s.y + beta * p.y,
                       alpha * s.z + beta * p.z, alpha * s.w + beta * p.w);
  }
}

// Transpose + zero-pad weight stack: dst[c][n][kp] = (kp < Krows) ? src[c][kp][n] : 0
__global__ void k_padT_w(const float* __restrict__ src, float* __restrict__ dst,
                         int Krows, int Kpad, int Ncols, int total) {
  int i = blockIdx.x * blockDim.x + threadIdx.x;
  if (i >= total) return;
  int kp = i % Kpad;
  int t  = i / Kpad;
  int n  = t % Ncols;
  int c  = t / Ncols;
  dst[i] = (kp < Krows) ? src[((size_t)c * Krows + kp) * Ncols + n] : 0.0f;
}

// ---------------- WMMA GEMM kernels (all dims compile-time) ----------------

// xf = relu(x[N,128] @ W_in[128,48] + b_in); BT is W_in transposed [48][128]
__global__ void k_gemm_in(const float* __restrict__ A, const float* __restrict__ BT,
                          const float* __restrict__ bias, float* __restrict__ out,
                          int tiles) {
  int wave = blockIdx.x * (blockDim.x >> 5) + (threadIdx.x >> 5);
  if (wave >= tiles) return;
  int lane = threadIdx.x & 31;
  const int nt = EMB / 16;
  int mt = wave / nt, ntile = wave - mt * nt;
  const float* At = A + (size_t)mt * 16 * FIN;
  const float* Bt = BT + (size_t)ntile * 16 * FIN;
  auto fa = [&](int m, int k) { return At[(size_t)m * FIN + k]; };
  auto fb = [&](int k, int n) { return Bt[(size_t)n * FIN + k]; };
  v8f c = {};
  c = wmma_tile<FIN>(fa, fb, c, lane);
  int n = ntile * 16 + (lane & 15);
  float bn = bias[n];
  size_t mbase = (size_t)mt * 16 + (size_t)(lane >> 4) * 8;
#pragma unroll
  for (int r = 0; r < 8; ++r) {
    float v = c[r] + bn;
    out[(mbase + r) * EMB + n] = v > 0.0f ? v : 0.0f;
  }
}

// acc (+)= T[N,48] @ W[48,48]; WT is transposed+padded [48][64]
__global__ void k_gemm_acc(const float* __restrict__ T, const float* __restrict__ WT,
                           float* __restrict__ acc, int accumulate, int tiles) {
  int wave = blockIdx.x * (blockDim.x >> 5) + (threadIdx.x >> 5);
  if (wave >= tiles) return;
  int lane = threadIdx.x & 31;
  const int nt = HID / 16;
  int mt = wave / nt, ntile = wave - mt * nt;
  const float* At = T + (size_t)mt * 16 * EMB;
  const float* Bt = WT + (size_t)ntile * 16 * 64;
  auto fa = [&](int m, int k) { return At[(size_t)m * EMB + k]; };
  auto fb = [&](int k, int n) { return Bt[(size_t)n * 64 + k]; };  // k < 64 padded
  int n = ntile * 16 + (lane & 15);
  size_t mbase = (size_t)mt * 16 + (size_t)(lane >> 4) * 8;
  v8f c = {};
  if (accumulate) {
#pragma unroll
    for (int r = 0; r < 8; ++r) c[r] = acc[(mbase + r) * HID + n];
  }
  c = wmma_tile<EMB>(fa, fb, c, lane);
#pragma unroll
  for (int r = 0; r < 8; ++r) acc[(mbase + r) * HID + n] = c[r];
}

// y = relu(acc + b_cheb) @ W_out + b_out; WT is transposed+padded [64][64]
__global__ void k_gemm_out(const float* __restrict__ acc, const float* __restrict__ bch,
                           const float* __restrict__ WT, const float* __restrict__ bo,
                           float* __restrict__ y, int tiles) {
  int wave = blockIdx.x * (blockDim.x >> 5) + (threadIdx.x >> 5);
  if (wave >= tiles) return;
  int lane = threadIdx.x & 31;
  const int nt = FOUT / 16;
  int mt = wave / nt, ntile = wave - mt * nt;
  const float* At = acc + (size_t)mt * 16 * HID;
  const float* Bt = WT + (size_t)ntile * 16 * 64;
  auto fa = [&](int m, int k) {
    float v = At[(size_t)m * HID + k] + bch[k];
    return v > 0.0f ? v : 0.0f;
  };
  auto fb = [&](int k, int n) { return Bt[(size_t)n * 64 + k]; };  // k < 64 padded
  v8f c = {};
  c = wmma_tile<HID>(fa, fb, c, lane);
  int n = ntile * 16 + (lane & 15);
  float bn = bo[n];
  size_t mbase = (size_t)mt * 16 + (size_t)(lane >> 4) * 8;
#pragma unroll
  for (int r = 0; r < 8; ++r) y[(mbase + r) * FOUT + n] = c[r] + bn;
}

// ---------------- host-side launch ----------------

extern "C" void kernel_launch(void* const* d_in, const int* in_sizes, int n_in,
                              void* d_out, int out_size, void* d_ws, size_t ws_size,
                              hipStream_t stream) {
  const float* x      = (const float*)d_in[0];
  const int*   ei     = (const int*)d_in[1];   // [2, E]
  const float* ew     = (const float*)d_in[2];
  const float* W_in   = (const float*)d_in[3];
  const float* b_in   = (const float*)d_in[4];
  const float* W_cheb = (const float*)d_in[5]; // [K, 48, 48]
  const float* b_cheb = (const float*)d_in[6];
  const float* W_out  = (const float*)d_in[7];
  const float* b_out  = (const float*)d_in[8];

  const int* row = ei;
  const int* col = ei + NEDGES;

  float* y  = (float*)d_out;               // [N, 64]
  float* xf = y + (size_t)NNODES * FOUT;   // [N, 48] second tuple output; doubles as T0

  float* ws    = (float*)d_ws;
  float* deg   = ws;                                  // N
  float* dinv  = deg + NNODES;                        // N
  float* norm  = dinv + NNODES;                       // E
  float* S     = norm + NEDGES;                       // N*48
  float* acc   = S + (size_t)NNODES * EMB;            // N*48
  float* tb0   = acc + (size_t)NNODES * EMB;          // 3 rotating T buffers
  float* tb1   = tb0 + (size_t)NNODES * EMB;
  float* tb2   = tb1 + (size_t)NNODES * EMB;
  float* WchT  = tb2 + (size_t)NNODES * EMB;          // 7 * 48 * 64  ([c][n][kpad])
  float* WoutT = WchT + (size_t)KORD * HID * 64;      // 64 * 64     ([n][kpad])
  float* WinT  = WoutT + (size_t)FOUT * 64;           // 48 * 128    ([n][k])
  float* Tbuf[3] = {tb0, tb1, tb2};

  const int TPB = 256;

  // transpose (+ zero-pad to K=64) all weights -> contiguous-K B operands
  {
    int tot = KORD * HID * 64;
    k_padT_w<<<(tot + TPB - 1) / TPB, TPB, 0, stream>>>(W_cheb, WchT, EMB, 64, HID, tot);
    tot = FOUT * 64;
    k_padT_w<<<(tot + TPB - 1) / TPB, TPB, 0, stream>>>(W_out, WoutT, HID, 64, FOUT, tot);
    tot = EMB * FIN;
    k_padT_w<<<(tot + TPB - 1) / TPB, TPB, 0, stream>>>(W_in, WinT, FIN, FIN, EMB, tot);
  }

  // normalization
  hipMemsetAsync(deg, 0, (size_t)NNODES * sizeof(float), stream);
  k_deg<<<(NEDGES + TPB - 1) / TPB, TPB, 0, stream>>>(row, ew, deg);
  k_dinv<<<(NNODES + TPB - 1) / TPB, TPB, 0, stream>>>(deg, dinv);
  k_norm<<<(NEDGES + TPB - 1) / TPB, TPB, 0, stream>>>(row, col, ew, dinv, norm);

  // xf = relu(x @ W_in + b_in)
  {
    int tiles = (NNODES / 16) * (EMB / 16);
    k_gemm_in<<<(tiles + 7) / 8, TPB, 0, stream>>>(x, WinT, b_in, xf, tiles);
  }

  const int tilesC = (NNODES / 16) * (HID / 16);
  const int blksC  = (tilesC + 7) / 8;

  // acc = T0 @ W_cheb[0]
  k_gemm_acc<<<blksC, TPB, 0, stream>>>(xf, WchT, acc, 0, tilesC);

  const float* Tprev2 = xf;
  const float* Tprev1 = xf;
  for (int k = 1; k < KORD; ++k) {
    float* Tcur = Tbuf[(k - 1) % 3];
    hipMemsetAsync(S, 0, (size_t)NNODES * EMB * sizeof(float), stream);
    k_prop<<<(NEDGES * 12 + TPB - 1) / TPB, TPB, 0, stream>>>(row, col, norm, Tprev1, S);
    float alpha = (k == 1) ? 1.0f : 2.0f;
    float beta  = (k == 1) ? 0.0f : -1.0f;
    k_combine<<<(NNODES * EMB / 4 + TPB - 1) / TPB, TPB, 0, stream>>>(
        (const float4*)S, (const float4*)Tprev2, (float4*)Tcur, alpha, beta);
    k_gemm_acc<<<blksC, TPB, 0, stream>>>(Tcur, WchT + (size_t)k * HID * 64, acc,
                                          1, tilesC);
    Tprev2 = Tprev1;
    Tprev1 = Tcur;
  }

  // y = relu(acc + b_cheb) @ W_out + b_out
  {
    int tiles = (NNODES / 16) * (FOUT / 16);
    k_gemm_out<<<(tiles + 7) / 8, TPB, 0, stream>>>(acc, b_cheb, WoutT, b_out, y, tiles);
  }
}
